// TransformerBlock_34729105555459
// MI455X (gfx1250) — matrix operand order repaired and checked by reading, not compiled, not run on hardware
//
#include <hip/hip_runtime.h>
#include <hip/hip_bf16.h>

// ---------------------------------------------------------------------------
// Types / helpers
// ---------------------------------------------------------------------------
typedef __attribute__((ext_vector_type(16))) __bf16 v16bf;
typedef __attribute__((ext_vector_type(8)))  float  v8f;
typedef __attribute__((ext_vector_type(4)))  unsigned int v4u;
typedef __attribute__((ext_vector_type(8)))  int    v8i;
typedef __attribute__((ext_vector_type(4)))  int    v4i;

union Frag16 {
    v16bf v;
    unsigned int u[8];
    unsigned short h[16];
};

__device__ __forceinline__ unsigned short f2bf(float f) {
    unsigned int u = __float_as_uint(f);
    u += 0x7FFFu + ((u >> 16) & 1u);          // round-to-nearest-even
    return (unsigned short)(u >> 16);
}

__device__ __forceinline__ v8f wmma_bf16(v16bf a, v16bf b, v8f c) {
    // (neg_a, A, neg_b, B, c_mod, C, reuse_a, reuse_b)
    return __builtin_amdgcn_wmma_f32_16x16x32_bf16(false, a, false, b, (short)0, c,
                                                   false, false);
}

// A-fragment gather (16-bit A 16x32 layout): bf16-pair dword loads
__device__ __forceinline__ void loadA(Frag16& f, const unsigned short* p, int half) {
#pragma unroll
    for (int q = 0; q < 8; ++q) {
        int e  = 2 * q;
        int kk = e + half * 16;
        f.u[q] = *(const unsigned int*)(p + kk);
    }
}

// Problem constants
#define SS   2048
#define BB   4
#define DD   512
#define HH   8
#define KK   64
#define FF   2048
#define MM   (SS * BB)   // 8192 rows of the "token" matrix

// ---------------------------------------------------------------------------
// Tensor Data Mover support (gfx1250). Guarded: falls back to direct loads.
// ---------------------------------------------------------------------------
#if defined(__has_builtin)
#  if __has_builtin(__builtin_amdgcn_tensor_load_to_lds) && \
      __has_builtin(__builtin_amdgcn_s_wait_tensorcnt)
#    define HAVE_TDM 1
#  endif
#endif
#ifndef HAVE_TDM
#  define HAVE_TDM 0
#endif

#if HAVE_TDM
// 2-D TDM copy: 2 rows x 2048 bf16 elements, row stride in elements.
// D# fields per CDNA5 ISA (#8.3 group0, #8.4 group1).
__device__ __forceinline__ void tdm_load_2x2048(unsigned lds_byte_off,
                                                const unsigned short* gsrc,
                                                long long row_stride_elems) {
    unsigned long long ga = (unsigned long long)(const void*)gsrc;
    v4u g0;
    g0.x = 1u;                                          // count=1 (user D#)
    g0.y = lds_byte_off;                                // lds_addr
    g0.z = (unsigned)(ga & 0xFFFFFFFFull);              // global_addr[31:0]
    g0.w = (unsigned)((ga >> 32) & 0x1FFFFFFull) | (2u << 30); // [56:32], type=2
    v8i g1;
    g1[0] = 0x00010000;                                 // data_size = 2 bytes
    g1[1] = (int)(2048u << 16);                         // tensor_dim0[15:0]=2048
    g1[2] = (int)(2u << 16);                            // dim0[31:16]=0 | tensor_dim1=2
    g1[3] = (int)(2048u << 16);                         // tile_dim0 = 2048
    g1[4] = 2;                                          // tile_dim1 = 2
    g1[5] = (int)(row_stride_elems & 0xFFFFFFFFll);     // dim0_stride[31:0]
    g1[6] = (int)((row_stride_elems >> 32) & 0xFFFFll); // dim0_stride[47:32]
    g1[7] = 0;
    v4i z4 = {0, 0, 0, 0};
#if defined(__clang_major__) && (__clang_major__ >= 23)
    v8i z8 = {0, 0, 0, 0, 0, 0, 0, 0};
    __builtin_amdgcn_tensor_load_to_lds(g0, g1, z4, z4, z8, 0);
#else
    __builtin_amdgcn_tensor_load_to_lds(g0, g1, z4, z4, 0);
#endif
}
__device__ __forceinline__ unsigned lds_offset_of(const void* p) {
    return (unsigned)(unsigned long long)p;             // low 32 bits = LDS addr
}
#endif

// ---------------------------------------------------------------------------
// fp32 -> bf16 elementwise convert
// ---------------------------------------------------------------------------
__global__ void k_f32_to_bf16(const float* __restrict__ src,
                              unsigned short* __restrict__ dst, int n) {
    int i = blockIdx.x * blockDim.x + threadIdx.x;
    if (i < n) dst[i] = f2bf(src[i]);
}

// ---------------------------------------------------------------------------
// Pack weight W[N][Kd] (row major fp32) into WMMA B-fragment-swizzled bf16.
// B = W^T, i.e. B[k][n] = W[n][k].
// B-fragment k order chosen here (contiguous, matches loadA): lane = half*16+lm -> col n = lm,
// element e -> k = half*16 + e.  Linear: ((kt*(N/16)+nt)*32+lane)*16+e
// ---------------------------------------------------------------------------
__global__ void k_pack_w(const float* __restrict__ W,
                         unsigned short* __restrict__ Bpk, int N, int Kd) {
    int i = blockIdx.x * blockDim.x + threadIdx.x;
    if (i >= N * Kd) return;
    int e    = i & 15;
    int lane = (i >> 4) & 31;
    int t    = i >> 9;              // kt*(N/16)+nt
    int ntls = N >> 4;
    int nt   = t % ntls;
    int kt   = t / ntls;
    int n    = nt * 16 + (lane & 15);
    int k    = kt * 32 + (lane >> 4) * 16 + e;
    Bpk[i] = f2bf(W[(size_t)n * Kd + k]);
}

// ---------------------------------------------------------------------------
// Generic WMMA bf16 GEMM:  C[M][N] = A[M][Kd] * Bpk  (+bias) (relu?)
// One wave -> 16(M) x 64(N) tile (4 accumulators), block = 8 waves in M.
// B panels staged by the Tensor Data Mover: one 2-D TDM tile = 2 k-steps
// (2 x 4KB, row stride ntls*512 elems), triple buffered in LDS so a single
// workgroup barrier per chunk suffices:
//   [wave0: s_wait_tensorcnt<=1 -> chunk c landed] ; barrier ;
//   [wave0: issue chunk c+2 -> buf (c+2)%3 (readers of it passed barrier)] ;
//   all waves: 16x ds_load_b128 + 8x WMMA on buf c%3.
// A fragments stay in registers, software pipelined one chunk ahead.
// ---------------------------------------------------------------------------
__global__ void k_gemm(const unsigned short* __restrict__ A,
                       const unsigned short* __restrict__ Bpk,
                       const float* __restrict__ bias,
                       float* __restrict__ Cf,
                       unsigned short* __restrict__ Cb,
                       int M, int N, int Kd, int relu) {
    const int lane = threadIdx.x & 31;
    const int wave = threadIdx.x >> 5;
    const int half = lane >> 4;
    const int lm   = lane & 15;
    const int m0   = (blockIdx.y * 8 + wave) * 16;
    const int n0   = blockIdx.x * 64;
    const int ntls = N >> 4;
    const int nt0  = n0 >> 4;
    const int NC   = Kd >> 6;                 // 64-wide k-chunks

    v8f acc[4] = {};
    const unsigned short* ap = A + (size_t)(m0 + lm) * Kd;

#if HAVE_TDM
    __shared__ unsigned short bsm[3][4096];   // 3 x 8KB B panels
    const long long bstride = (long long)ntls * 512;
    if (wave == 0) {                          // prologue: chunks 0 and 1
        tdm_load_2x2048(lds_offset_of(&bsm[0][0]),
                        Bpk + (size_t)nt0 * 512, bstride);
        if (NC > 1)
            tdm_load_2x2048(lds_offset_of(&bsm[1][0]),
                            Bpk + ((size_t)2 * ntls + nt0) * 512, bstride);
    }
#endif
    // A fragments for chunk 0 (k-steps 0 and 32)
    Frag16 af0, af1;
    loadA(af0, ap, half);
    loadA(af1, ap + 32, half);

    for (int c = 0; c < NC; ++c) {
#if HAVE_TDM
        if (wave == 0) {
            if (c + 1 < NC) __builtin_amdgcn_s_wait_tensorcnt(1);
            else            __builtin_amdgcn_s_wait_tensorcnt(0);
        }
        __syncthreads();                      // panel c visible; buf (c+2)%3 free
        if (wave == 0 && c + 2 < NC)
            tdm_load_2x2048(lds_offset_of(&bsm[(c + 2) % 3][0]),
                            Bpk + ((size_t)(2 * (c + 2)) * ntls + nt0) * 512,
                            bstride);
#endif
        // pipeline: A fragments for chunk c+1 while WMMAs run
        const int kn = (c + 1 < NC) ? (c + 1) * 64 : 0;
        __builtin_prefetch(ap + kn, 0, 1);
        Frag16 afn0, afn1;
        loadA(afn0, ap + kn, half);
        loadA(afn1, ap + kn + 32, half);

        // batch all B fragments (both k-steps), then all WMMAs
        Frag16 bf_[8];
#pragma unroll
        for (int s = 0; s < 2; ++s)
#pragma unroll
            for (int nb = 0; nb < 4; ++nb) {
#if HAVE_TDM
                bf_[s * 4 + nb].v =
                    *(const v16bf*)(&bsm[c % 3][(s * 128 + nb * 32 + lane) * 16]);
#else
                bf_[s * 4 + nb].v =
                    *(const v16bf*)(Bpk + ((size_t)((2 * c + s) * ntls + nt0 + nb) * 32 + lane) * 16);
#endif
            }
#pragma unroll
        for (int nb = 0; nb < 4; ++nb)
            acc[nb] = wmma_bf16(af0.v, bf_[nb].v, acc[nb]);
#pragma unroll
        for (int nb = 0; nb < 4; ++nb)
            acc[nb] = wmma_bf16(af1.v, bf_[4 + nb].v, acc[nb]);
        af0 = afn0;
        af1 = afn1;
    }

    // ---- epilogue: C layout lane->col n=lm, VGPR r -> row r + half*8
#pragma unroll
    for (int nb = 0; nb < 4; ++nb) {
#pragma unroll
        for (int r = 0; r < 8; ++r) {
            int m = m0 + r + half * 8;
            int n = n0 + nb * 16 + lm;
            float v = acc[nb][r];
            if (bias) v += bias[n];
            if (relu) v = v > 0.0f ? v : 0.0f;
            size_t idx = (size_t)m * N + n;
            if (Cf) Cf[idx] = v;
            if (Cb) Cb[idx] = f2bf(v);
        }
    }
}

// ---------------------------------------------------------------------------
// Attention re-pack kernels (plain [8192,512] bf16 -> per-head WMMA fragments)
// head n = b*H + h ; per-head Q_n[s][kk] = Qb[(s*4+b)*512 + h*64 + kk]
// ---------------------------------------------------------------------------
__global__ void k_pack_q(const unsigned short* __restrict__ Qb,
                         unsigned short* __restrict__ Qa) {
    int i = blockIdx.x * blockDim.x + threadIdx.x;   // 32*128*2*32*16 = 4M
    if (i >= 32 * SS * KK) return;
    int e    = i & 15;
    int lane = (i >> 4) & 31;
    int t    = i >> 9;          // (n*128+mt)*2+kt
    int kt   = t & 1;
    int mt   = (t >> 1) & 127;
    int n    = t >> 8;
    int half = lane >> 4, lm = lane & 15;
    int s  = mt * 16 + lm;
    int kk = kt * 32 + e + half * 16;   // A-fragment k map
    int b  = n >> 3, h = n & 7;
    Qa[i] = Qb[(size_t)(s * 4 + b) * 512 + h * 64 + kk];
}

__global__ void k_pack_kT(const unsigned short* __restrict__ Kb,
                          unsigned short* __restrict__ Kp) {
    int i = blockIdx.x * blockDim.x + threadIdx.x;
    if (i >= 32 * SS * KK) return;
    int e    = i & 15;
    int lane = (i >> 4) & 31;
    int t    = i >> 9;          // (n*2+kt)*128+nt
    int nt   = t & 127;
    int u    = t >> 7;
    int kt   = u & 1;
    int n    = u >> 1;
    int half = lane >> 4, lm = lane & 15;
    int tk = nt * 16 + lm;                 // key index (B col)
    int kk = kt * 32 + half * 16 + e;      // head-dim (B k) : B-fragment map
    int b  = n >> 3, h = n & 7;
    Kp[i] = Kb[(size_t)(tk * 4 + b) * 512 + h * 64 + kk];
}

__global__ void k_pack_v(const unsigned short* __restrict__ Vb,
                         unsigned short* __restrict__ Vp) {
    int i = blockIdx.x * blockDim.x + threadIdx.x;
    if (i >= 32 * SS * KK) return;
    int e    = i & 15;
    int lane = (i >> 4) & 31;
    int t    = i >> 9;          // (n*64+kt)*4+nt
    int nt   = t & 3;
    int u    = t >> 2;
    int kt   = u & 63;
    int n    = u >> 6;
    int half = lane >> 4, lm = lane & 15;
    int tk = kt * 32 + half * 16 + e;      // key index (B k)
    int kk = nt * 16 + lm;                 // head-dim (B col)
    int b  = n >> 3, h = n & 7;
    Vp[i] = Vb[(size_t)(tk * 4 + b) * 512 + h * 64 + kk];
}

// ---------------------------------------------------------------------------
// Flash attention: one wave owns a 16-query tile of one head.
// logits=(QK^T+mask)/8, online softmax fp32, P via LDS C->A transpose, P*V WMMA.
// ---------------------------------------------------------------------------
__global__ void k_attn(const unsigned short* __restrict__ Qa,
                       const unsigned short* __restrict__ Kp,
                       const unsigned short* __restrict__ Vp,
                       const float* __restrict__ mask,
                       unsigned short* __restrict__ Ub) {
    __shared__ unsigned short plds[8][16 * 32];      // per-wave P tile
    const int lane = threadIdx.x & 31;
    const int wave = threadIdx.x >> 5;
    const int half = lane >> 4;
    const int lm   = lane & 15;
    const int n    = blockIdx.y;                     // head
    const int mt   = blockIdx.x * 8 + wave;          // 16-query tile
    const int b    = n >> 3, h = n & 7;

    // Q fragments for both k-steps (contiguous 32B per lane)
    v16bf qf[2];
#pragma unroll
    for (int kt = 0; kt < 2; ++kt)
        qf[kt] = *(const v16bf*)(Qa + (((size_t)(n * 128 + mt) * 2 + kt) * 32 + lane) * 16);

    v8f o[4] = {};
    float mst[8], lst[8];
#pragma unroll
    for (int r = 0; r < 8; ++r) { mst[r] = -1e30f; lst[r] = 0.0f; }

    for (int j = 0; j < SS / 32; ++j) {              // 32-key chunks
        v8f s0 = {}, s1 = {};
#pragma unroll
        for (int kt = 0; kt < 2; ++kt) {
            v16bf b0 = *(const v16bf*)(Kp + (((size_t)(n * 2 + kt) * 128 + (2 * j + 0)) * 32 + lane) * 16);
            v16bf b1 = *(const v16bf*)(Kp + (((size_t)(n * 2 + kt) * 128 + (2 * j + 1)) * 32 + lane) * 16);
            s0 = wmma_bf16(qf[kt], b0, s0);
            s1 = wmma_bf16(qf[kt], b1, s1);
        }

        // (x + mask) / sqrt(K); C layout: col = tile*16+lm, row = r+half*8
        float sv[2][8];
#pragma unroll
        for (int tile = 0; tile < 2; ++tile)
#pragma unroll
            for (int r = 0; r < 8; ++r) {
                int mrow = mt * 16 + r + half * 8;
                int tcol = j * 32 + tile * 16 + lm;
                float x  = tile ? s1[r] : s0[r];
                sv[tile][r] = (x + mask[(size_t)mrow * SS + tcol]) * 0.125f;
            }

        // per-row max across the 16 key-lanes
        float rmx[8];
#pragma unroll
        for (int r = 0; r < 8; ++r) rmx[r] = fmaxf(sv[0][r], sv[1][r]);
#pragma unroll
        for (int d = 1; d < 16; d <<= 1)
#pragma unroll
            for (int r = 0; r < 8; ++r) rmx[r] = fmaxf(rmx[r], __shfl_xor(rmx[r], d, 32));

        float alpha[8], psum[8];
#pragma unroll
        for (int r = 0; r < 8; ++r) {
            float nm = fmaxf(mst[r], rmx[r]);
            alpha[r] = __expf(mst[r] - nm);
            mst[r]   = nm;
            psum[r]  = 0.0f;
        }

        // P = exp(s - m), write bf16 tile to LDS in [row][key] order
#pragma unroll
        for (int tile = 0; tile < 2; ++tile)
#pragma unroll
            for (int r = 0; r < 8; ++r) {
                float p = __expf(sv[tile][r] - mst[r]);
                psum[r] += p;
                plds[wave][(r + half * 8) * 32 + tile * 16 + lm] = f2bf(p);
            }
#pragma unroll
        for (int d = 1; d < 16; d <<= 1)
#pragma unroll
            for (int r = 0; r < 8; ++r) psum[r] += __shfl_xor(psum[r], d, 32);
#pragma unroll
        for (int r = 0; r < 8; ++r) lst[r] = lst[r] * alpha[r] + psum[r];

        // rescale O accumulators
#pragma unroll
        for (int nb = 0; nb < 4; ++nb)
#pragma unroll
            for (int r = 0; r < 8; ++r) o[nb][r] *= alpha[r];

        // wave-internal DS RAW: stores above, gathered loads below
        asm volatile("s_wait_dscnt 0" ::: "memory");

        // read P back as A-fragment (row = lm, k map = A layout)
        Frag16 pf;
        const unsigned short* lp = &plds[wave][lm * 32];
#pragma unroll
        for (int p = 0; p < 8; ++p) {
            int e  = 2 * p;
            int kk = e + half * 16;
            pf.u[p] = *(const unsigned int*)(lp + kk);
        }

        // O += P * V
#pragma unroll
        for (int nb = 0; nb < 4; ++nb) {
            v16bf vf = *(const v16bf*)(Vp + (((size_t)(n * 64 + j) * 4 + nb) * 32 + lane) * 16);
            o[nb] = wmma_bf16(pf.v, vf, o[nb]);
        }
    }

    // finalize and scatter to plain [8192,512] bf16 at head columns
#pragma unroll
    for (int r = 0; r < 8; ++r) {
        float inv = 1.0f / lst[r];
        int srow  = mt * 16 + r + half * 8;
#pragma unroll
        for (int nb = 0; nb < 4; ++nb) {
            int col = h * 64 + nb * 16 + lm;
            Ub[(size_t)(srow * 4 + b) * 512 + col] = f2bf(o[nb][r] * inv);
        }
    }
}

// ---------------------------------------------------------------------------
// out = LayerNorm(a + b) * g + beta ; optional fp32 and bf16 outputs. D=512.
// ---------------------------------------------------------------------------
__global__ void k_add_ln(const float* __restrict__ a, const float* __restrict__ bb,
                         const float* __restrict__ g, const float* __restrict__ be,
                         float* __restrict__ outf, unsigned short* __restrict__ outb) {
    __shared__ float ssum[8], ssq[8];
    const int row = blockIdx.x;
    const int tid = threadIdx.x;
    const size_t base = (size_t)row * 512;

    float v0 = a[base + tid]       + bb[base + tid];
    float v1 = a[base + 256 + tid] + bb[base + 256 + tid];
    float s = v0 + v1, q = v0 * v0 + v1 * v1;
#pragma unroll
    for (int d = 1; d < 32; d <<= 1) { s += __shfl_xor(s, d, 32); q += __shfl_xor(q, d, 32); }
    if ((tid & 31) == 0) { ssum[tid >> 5] = s; ssq[tid >> 5] = q; }
    __syncthreads();
    s = 0.0f; q = 0.0f;
#pragma unroll
    for (int i = 0; i < 8; ++i) { s += ssum[i]; q += ssq[i]; }
    float mu = s * (1.0f / 512.0f);
    float rs = rsqrtf(q * (1.0f / 512.0f) - mu * mu + 1e-5f);
    float y0 = (v0 - mu) * rs * g[tid]       + be[tid];
    float y1 = (v1 - mu) * rs * g[tid + 256] + be[tid + 256];
    if (outf) { outf[base + tid] = y0; outf[base + 256 + tid] = y1; }
    if (outb) { outb[base + tid] = f2bf(y0); outb[base + 256 + tid] = f2bf(y1); }
}

// ---------------------------------------------------------------------------
// Host orchestration
// ---------------------------------------------------------------------------
extern "C" void kernel_launch(void* const* d_in, const int* in_sizes, int n_in,
                              void* d_out, int out_size, void* d_ws, size_t ws_size,
                              hipStream_t stream) {
    const float* x    = (const float*)d_in[0];
    const float* mask = (const float*)d_in[1];
    const float* wq   = (const float*)d_in[2];
    const float* wk   = (const float*)d_in[3];
    const float* wv   = (const float*)d_in[4];
    const float* wc   = (const float*)d_in[5];
    const float* w1   = (const float*)d_in[6];
    const float* b1   = (const float*)d_in[7];
    const float* w2   = (const float*)d_in[8];
    const float* b2   = (const float*)d_in[9];
    const float* g1   = (const float*)d_in[10];
    const float* be1  = (const float*)d_in[11];
    const float* g2   = (const float*)d_in[12];
    const float* be2  = (const float*)d_in[13];

    char* ws = (char*)d_ws;
    const size_t MB = 1ull << 20;
    unsigned short* wqP = (unsigned short*)(ws + 0 * MB);
    unsigned short* wkP = (unsigned short*)(ws + 0 * MB + 512 * 1024);
    unsigned short* wvP = (unsigned short*)(ws + 1 * MB);
    unsigned short* wcP = (unsigned short*)(ws + 1 * MB + 512 * 1024);
    unsigned short* w1P = (unsigned short*)(ws + 2 * MB);
    unsigned short* w2P = (unsigned short*)(ws + 4 * MB);
    unsigned short* Xb  = (unsigned short*)(ws + 6 * MB);
    unsigned short* Qb  = (unsigned short*)(ws + 14 * MB);
    unsigned short* Kb  = (unsigned short*)(ws + 22 * MB);
    unsigned short* Vb  = (unsigned short*)(ws + 30 * MB);
    unsigned short* Qa  = (unsigned short*)(ws + 38 * MB);
    unsigned short* Kp  = (unsigned short*)(ws + 46 * MB);
    unsigned short* Vp  = (unsigned short*)(ws + 54 * MB);
    unsigned short* Ub  = (unsigned short*)(ws + 62 * MB);
    float*          Cp  = (float*)(ws + 70 * MB);
    unsigned short* U1b = (unsigned short*)(ws + 86 * MB);
    float*          U1f = (float*)(ws + 94 * MB);
    unsigned short* Zb  = (unsigned short*)(ws + 14 * MB);  // alias (Q/K/V dead)
    float*          Ff  = (float*)(ws + 46 * MB);           // alias (Kp/Vp dead)

    // 1) activations + weights to bf16 (weights into B-fragment swizzle)
    k_f32_to_bf16<<<(MM * DD) / 256, 256, 0, stream>>>(x, Xb, MM * DD);
    k_pack_w<<<(512 * 512) / 256, 256, 0, stream>>>(wq, wqP, 512, 512);
    k_pack_w<<<(512 * 512) / 256, 256, 0, stream>>>(wk, wkP, 512, 512);
    k_pack_w<<<(512 * 512) / 256, 256, 0, stream>>>(wv, wvP, 512, 512);
    k_pack_w<<<(512 * 512) / 256, 256, 0, stream>>>(wc, wcP, 512, 512);
    k_pack_w<<<(2048 * 512) / 256, 256, 0, stream>>>(w1, w1P, 2048, 512);
    k_pack_w<<<(512 * 2048) / 256, 256, 0, stream>>>(w2, w2P, 512, 2048);

    // 2) QKV projections
    dim3 gqkv(512 / 64, MM / 128);
    k_gemm<<<gqkv, 256, 0, stream>>>(Xb, wqP, nullptr, nullptr, Qb, MM, 512, 512, 0);
    k_gemm<<<gqkv, 256, 0, stream>>>(Xb, wkP, nullptr, nullptr, Kb, MM, 512, 512, 0);
    k_gemm<<<gqkv, 256, 0, stream>>>(Xb, wvP, nullptr, nullptr, Vb, MM, 512, 512, 0);

    // 3) per-head fragment re-pack
    int npk = 32 * SS * KK;
    k_pack_q <<<npk / 256, 256, 0, stream>>>(Qb, Qa);
    k_pack_kT<<<npk / 256, 256, 0, stream>>>(Kb, Kp);
    k_pack_v <<<npk / 256, 256, 0, stream>>>(Vb, Vp);

    // 4) attention (32 heads x 128 query tiles; 8 tiles per block)
    k_attn<<<dim3(16, 32), 256, 0, stream>>>(Qa, Kp, Vp, mask, Ub);

    // 5) output projection + residual + LN1
    k_gemm<<<gqkv, 256, 0, stream>>>(Ub, wcP, nullptr, Cp, nullptr, MM, 512, 512, 0);
    k_add_ln<<<MM, 256, 0, stream>>>(x, Cp, g1, be1, U1f, U1b);

    // 6) FFN
    k_gemm<<<dim3(2048 / 64, MM / 128), 256, 0, stream>>>(U1b, w1P, b1, nullptr, Zb,
                                                          MM, 2048, 512, 1);
    k_gemm<<<gqkv, 256, 0, stream>>>(Zb, w2P, b2, Ff, nullptr, MM, 512, 2048, 0);

    // 7) residual + LN2 -> output
    k_add_ln<<<MM, 256, 0, stream>>>(U1f, Ff, g2, be2, (float*)d_out, nullptr);
}
